// QTAttA_55602646614237
// MI455X (gfx1250) — matrix operand order repaired and checked by reading, not compiled, not run on hardware
//
#include <hip/hip_runtime.h>
#include <math.h>

// Problem constants (from reference)
#define Bq      2
#define Nq      2048
#define IN_DIMq 256
#define DIMq    256
#define Hq      8
#define DHq     32
#define TOPKq   4
#define ROWSq   (Bq * Nq)        // 4096
#define QKV_O   (3 * DIMq)       // 768

typedef __attribute__((ext_vector_type(16))) __bf16 v16bf;
typedef __attribute__((ext_vector_type(8)))  float  v8f;

union FragBF {
    v16bf v;
    uint4 q[2];
    unsigned short e[16];
};

__device__ __forceinline__ unsigned short f2bf(float f) {
    union { float f; unsigned u; } x; x.f = f;
    unsigned u = x.u;
    u += 0x7FFFu + ((u >> 16) & 1u);   // round-to-nearest-even
    return (unsigned short)(u >> 16);
}

// ---------------------------------------------------------------------------
// LayerNorm: one wave (32 lanes) per row of 256 f32 -> bf16
// ---------------------------------------------------------------------------
__global__ void qta_ln_kernel(const float* __restrict__ x,
                              const float* __restrict__ gamma,
                              const float* __restrict__ beta,
                              unsigned short* __restrict__ xn) {
    int wave = (blockIdx.x * blockDim.x + threadIdx.x) >> 5;
    int lane = threadIdx.x & 31;
    if (wave >= ROWSq) return;

    const float* row = x + (size_t)wave * IN_DIMq;
    float4 a = *(const float4*)(row + lane * 8);
    float4 b = *(const float4*)(row + lane * 8 + 4);
    float vals[8] = {a.x, a.y, a.z, a.w, b.x, b.y, b.z, b.w};

    float s = 0.f, s2 = 0.f;
#pragma unroll
    for (int i = 0; i < 8; i++) { s += vals[i]; s2 += vals[i] * vals[i]; }
#pragma unroll
    for (int m = 16; m >= 1; m >>= 1) {
        s  += __shfl_xor(s,  m, 32);
        s2 += __shfl_xor(s2, m, 32);
    }
    float mu  = s * (1.f / 256.f);
    float var = s2 * (1.f / 256.f) - mu * mu;
    float inv = rsqrtf(var + 1e-5f);

    unsigned short tmp[8];
#pragma unroll
    for (int i = 0; i < 8; i++) {
        int c = lane * 8 + i;
        tmp[i] = f2bf((vals[i] - mu) * inv * gamma[c] + beta[c]);
    }
    // pack 8 bf16 -> one 16B store
    uint4 pk;
    pk.x = (unsigned)tmp[0] | ((unsigned)tmp[1] << 16);
    pk.y = (unsigned)tmp[2] | ((unsigned)tmp[3] << 16);
    pk.z = (unsigned)tmp[4] | ((unsigned)tmp[5] << 16);
    pk.w = (unsigned)tmp[6] | ((unsigned)tmp[7] << 16);
    *(uint4*)(xn + (size_t)wave * IN_DIMq + lane * 8) = pk;
}

// ---------------------------------------------------------------------------
// Weight f32 -> bf16 conversion
// ---------------------------------------------------------------------------
__global__ void qta_cvt_kernel(const float* __restrict__ wqkv,
                               const float* __restrict__ wproj,
                               unsigned short* __restrict__ wqkv_bf,
                               unsigned short* __restrict__ wproj_bf) {
    int i = blockIdx.x * blockDim.x + threadIdx.x;
    if (i < QKV_O * IN_DIMq) wqkv_bf[i] = f2bf(wqkv[i]);
    int j = i - QKV_O * IN_DIMq;
    if (j >= 0 && j < DIMq * DIMq) wproj_bf[j] = f2bf(wproj[j]);
}

// ---------------------------------------------------------------------------
// QKV GEMM: xn(4096x256 bf16) @ w_qkv^T -> q,k head-major, v^T, v f32
// One wave = one 16x16 output tile, K=256 in 8 wmma steps.
// A frag (16-bit 16x32): lane<16: row=lane, K 0..15; lane>=16: row=lane-16, K 16..31
// B frag (16-bit 32x16): lane: col=lane&15, K base = (lane<16?0:16), 16 contiguous K
// C frag: VGPR j: row = j + 8*(lane>=16), col = lane&15
// ---------------------------------------------------------------------------
__global__ void qta_qkv_gemm(const unsigned short* __restrict__ xn,
                             const unsigned short* __restrict__ wqkv,
                             unsigned short* __restrict__ qb,
                             unsigned short* __restrict__ kbuf,
                             unsigned short* __restrict__ vT,
                             float* __restrict__ vf) {
    int tile = blockIdx.x;            // 256 * 48 tiles
    int ot = tile % 48, mt = tile / 48;
    int m0 = mt * 16, o0 = ot * 16;
    int lane = threadIdx.x;
    int half = lane >> 4;
    int lc = lane & 15;

    v8f acc = {0.f, 0.f, 0.f, 0.f, 0.f, 0.f, 0.f, 0.f};
    const unsigned short* arow = xn   + (size_t)(m0 + lc) * IN_DIMq;
    const unsigned short* brow = wqkv + (size_t)(o0 + lc) * IN_DIMq;
    int ab = half ? 16 : 0;
    int bb = half ? 16 : 0;

#pragma unroll
    for (int k0 = 0; k0 < IN_DIMq; k0 += 32) {
        FragBF A, B;
        A.q[0] = *(const uint4*)(arow + k0 + ab);
        A.q[1] = *(const uint4*)(arow + k0 + ab + 8);
        B.q[0] = *(const uint4*)(brow + k0 + bb);
        B.q[1] = *(const uint4*)(brow + k0 + bb + 8);
        acc = __builtin_amdgcn_wmma_f32_16x16x32_bf16(false, A.v, false, B.v,
                                                      (short)0, acc, false, false);
    }

    int o   = o0 + lc;          // uniform matrix selector within a tile
    int mat = o >> 8;           // 0=q 1=k 2=v
    int oo  = o & 255;
    int h = oo >> 5, d = oo & 31;
#pragma unroll
    for (int j = 0; j < 8; j++) {
        int m = m0 + j + half * 8;
        int b = m >> 11, l = m & 2047;
        float val = acc[j];
        if (mat == 0) {
            qb[(((size_t)(b * Hq + h)) * Nq + l) * DHq + d] = f2bf(val);
        } else if (mat == 1) {
            kbuf[(((size_t)(b * Hq + h)) * Nq + l) * DHq + d] = f2bf(val);
        } else {
            vT[(((size_t)(b * Hq + h)) * DHq + d) * Nq + l] = f2bf(val);
            vf[(size_t)m * DIMq + oo] = val;
        }
    }
}

// ---------------------------------------------------------------------------
// Flash attention with running top-4 and top-4-masked output.
// One wave per (b, h, 16-query-row tile).  64 steps of 32 keys:
//   2 wmma for S, softmax+top4 via LDS, 2 wmma for P @ V^T.
// ---------------------------------------------------------------------------
__global__ void qta_attn_kernel(const unsigned short* __restrict__ qb,
                                const unsigned short* __restrict__ kbuf,
                                const unsigned short* __restrict__ vT,
                                const float* __restrict__ vf,
                                unsigned short* __restrict__ msg,
                                float* __restrict__ out_score,
                                int* __restrict__ out_idx) {
    __shared__ __attribute__((aligned(16))) float          ldsS[16][32];
    __shared__ __attribute__((aligned(16))) unsigned short ldsP[16][32];
    __shared__ float ldsC[16];
    __shared__ float ldsInv[16];
    __shared__ float ldsTP[16][TOPKq];
    __shared__ int   ldsTI[16][TOPKq];

    int tile = blockIdx.x;               // 2048 = B*H*(N/16)
    int lt = tile & 127;
    int h  = (tile >> 7) & 7;
    int b  = tile >> 10;
    int l0 = lt << 4;
    int lane = threadIdx.x;
    int half = lane >> 4;
    int lc = lane & 15;

    const float temp = 0.17677669529663687f;   // 1/sqrt(32)
    const float NEG_INF = -__builtin_inff();

    size_t headK  = ((size_t)(b * Hq + h)) * Nq * DHq;
    size_t headVT = ((size_t)(b * Hq + h)) * DHq * Nq;

    // Persistent Q A-fragment (16x32, single K step since DH=32)
    FragBF Aq;
    {
        const unsigned short* qrow = qb + headK + (size_t)(l0 + lc) * DHq;
        int ab = half ? 16 : 0;
        Aq.q[0] = *(const uint4*)(qrow + ab);
        Aq.q[1] = *(const uint4*)(qrow + ab + 8);
    }

    v8f acc0 = {0.f, 0.f, 0.f, 0.f, 0.f, 0.f, 0.f, 0.f};
    v8f acc1 = {0.f, 0.f, 0.f, 0.f, 0.f, 0.f, 0.f, 0.f};
    float m = NEG_INF, ssum = 0.f;
    float t0 = NEG_INF, t1 = NEG_INF, t2 = NEG_INF, t3 = NEG_INF;
    int   i0 = 0, i1 = 0, i2 = 0, i3 = 0;

    for (int s0 = 0; s0 < Nq; s0 += 32) {
        // ---- S = Q @ K^T over 32 key columns ----
        FragBF Bk0, Bk1;
        {
            int kb = half ? 16 : 0;
            const unsigned short* kr0 = kbuf + headK + (size_t)(s0 + lc) * DHq + kb;
            const unsigned short* kr1 = kbuf + headK + (size_t)(s0 + 16 + lc) * DHq + kb;
            Bk0.q[0] = *(const uint4*)(kr0);
            Bk0.q[1] = *(const uint4*)(kr0 + 8);
            Bk1.q[0] = *(const uint4*)(kr1);
            Bk1.q[1] = *(const uint4*)(kr1 + 8);
        }
        v8f sc0 = {0.f, 0.f, 0.f, 0.f, 0.f, 0.f, 0.f, 0.f};
        v8f sc1 = {0.f, 0.f, 0.f, 0.f, 0.f, 0.f, 0.f, 0.f};
        sc0 = __builtin_amdgcn_wmma_f32_16x16x32_bf16(false, Aq.v, false, Bk0.v,
                                                      (short)0, sc0, false, false);
        sc1 = __builtin_amdgcn_wmma_f32_16x16x32_bf16(false, Aq.v, false, Bk1.v,
                                                      (short)0, sc1, false, false);
#pragma unroll
        for (int j = 0; j < 8; j++) {
            int r = j + half * 8;
            ldsS[r][lc]      = temp * sc0[j];
            ldsS[r][16 + lc] = temp * sc1[j];
        }
        __syncthreads();

        // ---- row-owner lanes: flash softmax state + top-4 ----
        if (lane < 16) {
            int r = lane;
            float sv[32];
            float tmax = NEG_INF;
#pragma unroll
            for (int j = 0; j < 32; j++) { sv[j] = ldsS[r][j]; tmax = fmaxf(tmax, sv[j]); }
            float newm = fmaxf(m, tmax);
            float cf = __expf(m - newm);
            float psum = 0.f;
#pragma unroll
            for (int j = 0; j < 32; j++) {
                float p = __expf(sv[j] - newm);
                psum += p;
                ldsP[r][j] = f2bf(p);
                float s = sv[j];
                int idx = s0 + j;
                if (s > t0)      { t3=t2;i3=i2; t2=t1;i2=i1; t1=t0;i1=i0; t0=s;i0=idx; }
                else if (s > t1) { t3=t2;i3=i2; t2=t1;i2=i1; t1=s;i1=idx; }
                else if (s > t2) { t3=t2;i3=i2; t2=s;i2=idx; }
                else if (s > t3) { t3=s;i3=idx; }
            }
            ssum = ssum * cf + psum;
            m = newm;
            ldsC[r] = cf;
        }
        __syncthreads();

        // ---- rescale accumulator by per-row correction ----
#pragma unroll
        for (int j = 0; j < 8; j++) {
            float f = ldsC[j + half * 8];
            acc0[j] *= f;
            acc1[j] *= f;
        }

        // ---- O += P @ V^T ----
        FragBF Ap;
        {
            int ab = half ? 16 : 0;
            Ap.q[0] = *(const uint4*)&ldsP[lc][ab];
            Ap.q[1] = *(const uint4*)&ldsP[lc][ab + 8];
        }
        FragBF Bv0, Bv1;
        {
            int sb = s0 + (half ? 16 : 0);
            const unsigned short* v0 = vT + headVT + (size_t)lc * Nq + sb;
            const unsigned short* v1 = vT + headVT + (size_t)(16 + lc) * Nq + sb;
            Bv0.q[0] = *(const uint4*)(v0);
            Bv0.q[1] = *(const uint4*)(v0 + 8);
            Bv1.q[0] = *(const uint4*)(v1);
            Bv1.q[1] = *(const uint4*)(v1 + 8);
        }
        acc0 = __builtin_amdgcn_wmma_f32_16x16x32_bf16(false, Ap.v, false, Bv0.v,
                                                       (short)0, acc0, false, false);
        acc1 = __builtin_amdgcn_wmma_f32_16x16x32_bf16(false, Ap.v, false, Bv1.v,
                                                       (short)0, acc1, false, false);
        __syncthreads();
    }

    // ---- finalize: top-4 outputs + normalization constants ----
    if (lane < 16) {
        int r = lane;
        float inv = 1.f / ssum;
        ldsInv[r] = inv;
        float p0 = __expf(t0 - m) * inv;
        float p1 = __expf(t1 - m) * inv;
        float p2 = __expf(t2 - m) * inv;
        float p3 = __expf(t3 - m) * inv;
        ldsTP[r][0] = p0; ldsTP[r][1] = p1; ldsTP[r][2] = p2; ldsTP[r][3] = p3;
        ldsTI[r][0] = i0; ldsTI[r][1] = i1; ldsTI[r][2] = i2; ldsTI[r][3] = i3;
        int l = l0 + r;
        size_t obase = ((size_t)(b * Nq + l)) * (TOPKq * Hq);
        out_score[obase + 0 * Hq + h] = p0;
        out_score[obase + 1 * Hq + h] = p1;
        out_score[obase + 2 * Hq + h] = p2;
        out_score[obase + 3 * Hq + h] = p3;
        out_idx[obase + 0 * Hq + h] = i0;
        out_idx[obase + 1 * Hq + h] = i1;
        out_idx[obase + 2 * Hq + h] = i2;
        out_idx[obase + 3 * Hq + h] = i3;
    }
    __syncthreads();

    // stage O tile into LDS
#pragma unroll
    for (int j = 0; j < 8; j++) {
        int r = j + half * 8;
        ldsS[r][lc]      = acc0[j];
        ldsS[r][16 + lc] = acc1[j];
    }
    __syncthreads();

    // masked message = O/ssum - sum_i p_i * V[idx_i]
    {
        int r  = lane >> 1;
        int cb = (lane & 1) * 16;
        float inv = ldsInv[r];
        int l = l0 + r;
        size_t mrow = ((size_t)(b * Nq + l)) * DIMq + h * DHq + cb;
        float pp[4]; int id[4];
#pragma unroll
        for (int i = 0; i < 4; i++) { pp[i] = ldsTP[r][i]; id[i] = ldsTI[r][i]; }
#pragma unroll
        for (int c = 0; c < 16; c++) {
            float o = ldsS[r][cb + c] * inv;
#pragma unroll
            for (int i = 0; i < 4; i++)
                o -= pp[i] * vf[((size_t)(b * Nq + id[i])) * DIMq + h * DHq + cb + c];
            msg[mrow + c] = f2bf(o);
        }
    }
}

// ---------------------------------------------------------------------------
// Output projection + bias + residual: out = msg @ w_proj^T + b_proj + v_flat
// ---------------------------------------------------------------------------
__global__ void qta_proj_kernel(const unsigned short* __restrict__ msg,
                                const unsigned short* __restrict__ wproj,
                                const float* __restrict__ bproj,
                                const float* __restrict__ vf,
                                float* __restrict__ out) {
    int tile = blockIdx.x;            // 256 * 16 tiles
    int ot = tile & 15, mt = tile >> 4;
    int m0 = mt * 16, o0 = ot * 16;
    int lane = threadIdx.x;
    int half = lane >> 4;
    int lc = lane & 15;

    v8f acc = {0.f, 0.f, 0.f, 0.f, 0.f, 0.f, 0.f, 0.f};
    const unsigned short* arow = msg   + (size_t)(m0 + lc) * DIMq;
    const unsigned short* brow = wproj + (size_t)(o0 + lc) * DIMq;
    int ab = half ? 16 : 0;
    int bb = half ? 16 : 0;

#pragma unroll
    for (int k0 = 0; k0 < DIMq; k0 += 32) {
        FragBF A, B;
        A.q[0] = *(const uint4*)(arow + k0 + ab);
        A.q[1] = *(const uint4*)(arow + k0 + ab + 8);
        B.q[0] = *(const uint4*)(brow + k0 + bb);
        B.q[1] = *(const uint4*)(brow + k0 + bb + 8);
        acc = __builtin_amdgcn_wmma_f32_16x16x32_bf16(false, A.v, false, B.v,
                                                      (short)0, acc, false, false);
    }

    int o = o0 + lc;
    float bp = bproj[o];
#pragma unroll
    for (int j = 0; j < 8; j++) {
        int n = m0 + j + half * 8;
        out[(size_t)n * DIMq + o] = acc[j] + bp + vf[(size_t)n * DIMq + o];
    }
}

// ---------------------------------------------------------------------------
extern "C" void kernel_launch(void* const* d_in, const int* in_sizes, int n_in,
                              void* d_out, int out_size, void* d_ws, size_t ws_size,
                              hipStream_t stream) {
    const float* points = (const float*)d_in[0];
    const float* gamma  = (const float*)d_in[1];
    const float* beta   = (const float*)d_in[2];
    const float* w_qkv  = (const float*)d_in[3];
    const float* w_proj = (const float*)d_in[4];
    const float* b_proj = (const float*)d_in[5];

    // workspace carve-up (~14.5 MB, all chunks 256B-aligned)
    char* w = (char*)d_ws;
    unsigned short* xn       = (unsigned short*)w; w += (size_t)ROWSq * IN_DIMq * 2;
    unsigned short* wqkv_bf  = (unsigned short*)w; w += (size_t)QKV_O * IN_DIMq * 2;
    unsigned short* wproj_bf = (unsigned short*)w; w += (size_t)DIMq * DIMq * 2;
    unsigned short* qb       = (unsigned short*)w; w += (size_t)ROWSq * DIMq * 2;
    unsigned short* kbuf     = (unsigned short*)w; w += (size_t)ROWSq * DIMq * 2;
    unsigned short* vT       = (unsigned short*)w; w += (size_t)ROWSq * DIMq * 2;
    unsigned short* msg      = (unsigned short*)w; w += (size_t)ROWSq * DIMq * 2;
    float*          vf       = (float*)w;          w += (size_t)ROWSq * DIMq * 4;

    float* out_msg   = (float*)d_out;                      // 4096*256
    float* out_score = out_msg + (size_t)ROWSq * DIMq;     // 4096*4*8
    int*   out_idx   = (int*)(out_msg + (size_t)ROWSq * DIMq + (size_t)ROWSq * TOPKq * Hq);

    // 1) LayerNorm (1 wave/row, 4 waves/block)
    qta_ln_kernel<<<ROWSq / 4, 128, 0, stream>>>(points, gamma, beta, xn);

    // 2) weight conversion
    int cvt_n = QKV_O * IN_DIMq + DIMq * DIMq;
    qta_cvt_kernel<<<(cvt_n + 255) / 256, 256, 0, stream>>>(w_qkv, w_proj, wqkv_bf, wproj_bf);

    // 3) QKV GEMM: 256 x 48 tiles, 1 wave each
    qta_qkv_gemm<<<256 * 48, 32, 0, stream>>>(xn, wqkv_bf, qb, kbuf, vT, vf);

    // 4) flash attention + running top-4: B*H*(N/16) = 2048 waves
    qta_attn_kernel<<<Bq * Hq * (Nq / 16), 32, 0, stream>>>(qb, kbuf, vT, vf, msg,
                                                            out_score, out_idx);

    // 5) projection + bias + residual: 256 x 16 tiles
    qta_proj_kernel<<<256 * 16, 32, 0, stream>>>(msg, wproj_bf, b_proj, vf, out_msg);
}
